// MultiHeadAttention_22686017257524
// MI455X (gfx1250) — compile-verified, matrix operand order checked by reading, not run on hardware
//
#include <hip/hip_runtime.h>

#define EMB 1024
#define HEADS 16
#define HEAD_DIM 64
#define BATCH 2
#define SEQ 2048
#define MTOT (BATCH * SEQ)   // 4096
#define QKVO (3 * EMB)       // 3072

typedef __attribute__((ext_vector_type(16))) __bf16       v16bf;
typedef __attribute__((ext_vector_type(8)))  float        v8f;
typedef __attribute__((ext_vector_type(4)))  unsigned int u32x4;
typedef __attribute__((ext_vector_type(4)))  int          i32x4;
typedef __attribute__((ext_vector_type(8)))  int          i32x8;
typedef unsigned short us16;

union frag_u { u32x4 q[2]; v16bf v; };

// fp32 -> bf16 round-to-nearest-even
__device__ __forceinline__ us16 f2bf(float f) {
  unsigned int u = __float_as_uint(f);
  u += 0x7FFFu + ((u >> 16) & 1u);
  return (us16)(u >> 16);
}

// 16x32 bf16 A-fragment (or symmetric 32x16 B-fragment) for
// v_wmma_f32_16x16x32_bf16. Halves [0..7] = K {k0..k0+7},
// halves [8..15] = K {k0+16..k0+23}.
__device__ __forceinline__ v16bf load_frag(const us16* p) {
  frag_u u;
  u.q[0] = *(const u32x4*)(p);
  u.q[1] = *(const u32x4*)(p + 16);
  return u.v;
}

#define WMMA_BF16(a, b, c) \
  __builtin_amdgcn_wmma_f32_16x16x32_bf16(false, (a), false, (b), (short)0, (c), false, false)

// ---- CDNA5 async global->LDS copy (ASYNCcnt, in-order completion)
__device__ __forceinline__ void async_b128(const us16* g, const us16* l) {
  // generic LDS pointer = {aperture_hi, lds_offset32}; low 32 bits are the DS address
  unsigned lds = (unsigned)(unsigned long long)l;
  asm volatile("global_load_async_to_lds_b128 %0, %1, off"
               :: "v"(lds), "v"(g) : "memory");
}
#define S_WAIT_ASYNC_0() asm volatile("s_wait_asynccnt 0x0" ::: "memory")
#define S_WAIT_ASYNC_4() asm volatile("s_wait_asynccnt 0x4" ::: "memory")

// ---- CDNA5 Tensor Data Mover: 2D tile (tile_rows x 32 bf16) global->LDS.
// D# group0/group1 per CDNA5 ISA; VADDR2/3 NULL
// (<=2D tensor). Row stride = EMB elements, data_size = 2 bytes.
__device__ __forceinline__ void tdm_load_tile(const us16* gtile, const us16* ltile,
                                              int tile_rows, int tensor_rows) {
  unsigned long long ga = (unsigned long long)gtile;
  unsigned lds = (unsigned)(unsigned long long)ltile;
  i32x4 g0;
  g0[0] = 1;                                   // count=1 (valid user descriptor)
  g0[1] = (int)lds;                            // lds_addr
  g0[2] = (int)(ga & 0xffffffffull);           // global_addr[31:0]
  g0[3] = (int)((ga >> 32) & 0x01ffffffull) |  // global_addr[56:32]
          (int)0x80000000;                     // type=2 ("image")
  i32x8 g1;
  g1[0] = 0x10000;                             // workgroup_mask=0, data_size=1 (2B)
  g1[1] = (EMB & 0xffff) << 16;                // tensor_dim0[15:0]
  g1[2] = ((EMB >> 16) & 0xffff) |             // tensor_dim0[31:16]
          ((tensor_rows & 0xffff) << 16);      // tensor_dim1[15:0]
  g1[3] = ((tensor_rows >> 16) & 0xffff) |     // tensor_dim1[31:16]
          (32 << 16);                          // tile_dim0 = 32
  g1[4] = tile_rows & 0xffff;                  // tile_dim1; tile_dim2=0
  g1[5] = EMB;                                 // tensor_dim0_stride[31:0]
  g1[6] = 0;                                   // stride[47:32]=0; dim1_stride lo
  g1[7] = 0;
  asm volatile("tensor_load_to_lds %0, %1" :: "s"(g0), "s"(g1) : "memory");
}

// ---------------------------------------------------------------- conversions
__global__ void cvt_f32_bf16(const float* __restrict__ src, us16* __restrict__ dst, int n) {
  int i = blockIdx.x * blockDim.x + threadIdx.x;
  if (i < n) dst[i] = f2bf(src[i]);
}

// ---------------------------------------------------------------- QKV GEMM
// 8 waves, 64(M) x 128(O) tile; A/B k-slabs TDM-staged into double-buffered
// LDS. qkv[m,o] = sum_k x[m,k]*qkv_w[o,k] + b[o]; scatter via o = h*192+d*3+c.
__global__ __launch_bounds__(256) void qkv_gemm(const us16* __restrict__ xb,
                                                const us16* __restrict__ wb,
                                                const float* __restrict__ bias,
                                                us16* __restrict__ Q,
                                                us16* __restrict__ Kt,
                                                us16* __restrict__ Vt) {
  __shared__ __align__(16) us16 Abuf[2][64 * 32];    // 8 KB
  __shared__ __align__(16) us16 Bbuf[2][128 * 32];   // 16 KB

  const int tid  = threadIdx.x;
  const int w    = tid >> 5, lane = tid & 31;
  const int rc   = lane & 15;
  const int ks   = (lane >> 4) << 3;
  const int half = lane >> 4;
  const int mi   = w & 3;        // m-subtile 0..3
  const int ng   = w >> 2;       // n-group 0..1 (64 cols each)
  const int o0   = blockIdx.x * 128;
  const int m0   = blockIdx.y * 64;

  auto stage = [&](int buf, int k0) {  // wave 0 only: 2 TDM descriptors
    tdm_load_tile(xb + (size_t)m0 * EMB + k0, &Abuf[buf][0], 64,  MTOT);
    tdm_load_tile(wb + (size_t)o0 * EMB + k0, &Bbuf[buf][0], 128, QKVO);
  };

  v8f acc[4] = {v8f{}, v8f{}, v8f{}, v8f{}};
  if (w == 0) stage(0, 0);
  for (int k0 = 0; k0 < EMB; k0 += 32) {
    const int cur = (k0 >> 5) & 1;
    if (w == 0) {
      if (k0 + 32 < EMB) { stage(cur ^ 1, k0 + 32); __builtin_amdgcn_s_wait_tensorcnt(2); }
      else               { __builtin_amdgcn_s_wait_tensorcnt(0); }
    }
    __syncthreads();
    v16bf a = load_frag(&Abuf[cur][(mi * 16 + rc) * 32 + ks]);
#pragma unroll
    for (int t = 0; t < 4; ++t) {
      v16bf b = load_frag(&Bbuf[cur][(ng * 64 + t * 16 + rc) * 32 + ks]);
      acc[t] = WMMA_BF16(a, b, acc[t]);
    }
    __syncthreads();
  }

#pragma unroll
  for (int t = 0; t < 4; ++t) {
#pragma unroll
    for (int j = 0; j < 8; ++j) {
      int m = m0 + mi * 16 + j + half * 8;
      int o = o0 + ng * 64 + t * 16 + rc;
      float v = acc[t][j] + bias[o];
      int b_ = m / SEQ, n_ = m % SEQ;
      int h  = o / (HEAD_DIM * 3);
      int r  = o % (HEAD_DIM * 3);
      int d  = r / 3, c = r % 3;
      us16 bv = f2bf(v);
      size_t bh = (size_t)(b_ * HEADS + h);
      if (c == 0)      Q [(bh * SEQ + n_) * HEAD_DIM + d] = bv;
      else if (c == 1) Kt[(bh * SEQ + n_) * HEAD_DIM + d] = bv;
      else             Vt[(bh * HEAD_DIM + d) * SEQ + n_] = bv;  // V transposed
    }
  }
}

// ---------------------------------------------------------------- attention
// 4 waves x 16 queries = 64 queries per block; K/V tiles (32 keys) shared via
// double-buffered async-LDS staging (ASYNCcnt path). Online softmax; 1/512.
__global__ __launch_bounds__(128) void attn(const us16* __restrict__ Q,
                                            const us16* __restrict__ K,
                                            const us16* __restrict__ Vt,
                                            us16* __restrict__ ctx) {
  __shared__ __align__(16) us16 Kbuf[2][32 * HEAD_DIM];  // 8 KB
  __shared__ __align__(16) us16 Vbuf[2][HEAD_DIM * 32];  // 8 KB
  __shared__ __align__(16) us16 Pbuf[4][16 * 32];        // 4 KB (per-wave P bounce)

  const int tid  = threadIdx.x;
  const int w    = tid >> 5, lane = tid & 31;
  const int rc   = lane & 15;
  const int ks   = (lane >> 4) << 3;
  const int half = lane >> 4;
  const int q0   = blockIdx.x * 64 + w * 16;
  const int bh   = blockIdx.y;

  const us16* Kb = K  + (size_t)bh * SEQ * HEAD_DIM;
  const us16* Vb = Vt + (size_t)bh * HEAD_DIM * SEQ;

  const us16* Qp = Q + ((size_t)bh * SEQ + q0 + rc) * HEAD_DIM + ks;
  const v16bf qa0 = load_frag(Qp);        // d = 0..31
  const v16bf qa1 = load_frag(Qp + 32);   // d = 32..63

  auto stageKV = [&](int buf, int kb) {
#pragma unroll
    for (int cc = 0; cc < 2; ++cc) { // K tile: 32 rows x 64 halves = 256 chunks
      int c = tid + cc * 128;
      int row = c >> 3, ko = (c & 7) << 3;
      async_b128(Kb + (size_t)(kb + row) * HEAD_DIM + ko, &Kbuf[buf][row * HEAD_DIM + ko]);
    }
#pragma unroll
    for (int cc = 0; cc < 2; ++cc) { // V tile: 64 rows x 32 halves = 256 chunks
      int c = tid + cc * 128;
      int row = c >> 2, ko = (c & 3) << 3;
      async_b128(Vb + (size_t)row * SEQ + kb + ko, &Vbuf[buf][row * 32 + ko]);
    }
  };

  v8f O[4] = {v8f{}, v8f{}, v8f{}, v8f{}};
  float mrow[8], lrow[8];
#pragma unroll
  for (int j = 0; j < 8; ++j) { mrow[j] = -1e30f; lrow[j] = 0.0f; }
  const float sc = 1.0f / 512.0f;

  stageKV(0, 0);
  for (int kb = 0; kb < SEQ; kb += 32) {
    const int cur = (kb >> 5) & 1;
    if (kb + 32 < SEQ) { stageKV(cur ^ 1, kb + 32); S_WAIT_ASYNC_4(); }
    else               { S_WAIT_ASYNC_0(); }
    __syncthreads();

    // ---- S = Q K^T for 32 keys (two 16x16 C tiles), fragments from LDS
    v8f sA = {}, sB = {};
    {
      v16bf b0 = load_frag(&Kbuf[cur][(rc) * HEAD_DIM + ks]);
      v16bf b1 = load_frag(&Kbuf[cur][(rc) * HEAD_DIM + 32 + ks]);
      sA = WMMA_BF16(qa0, b0, sA);
      sA = WMMA_BF16(qa1, b1, sA);
    }
    {
      v16bf b0 = load_frag(&Kbuf[cur][(16 + rc) * HEAD_DIM + ks]);
      v16bf b1 = load_frag(&Kbuf[cur][(16 + rc) * HEAD_DIM + 32 + ks]);
      sB = WMMA_BF16(qa0, b0, sB);
      sB = WMMA_BF16(qa1, b1, sB);
    }

    // ---- online softmax (row reductions stay inside each 16-lane half)
    float alpha[8];
#pragma unroll
    for (int j = 0; j < 8; ++j) {
      float sa = sA[j] * sc, sb = sB[j] * sc;
      float mx = fmaxf(sa, sb);
#pragma unroll
      for (int s = 1; s < 16; s <<= 1) mx = fmaxf(mx, __shfl_xor(mx, s, 32));
      float nm = fmaxf(mrow[j], mx);
      float al = __expf(mrow[j] - nm);
      float pa = __expf(sa - nm);
      float pb = __expf(sb - nm);
      float rs = pa + pb;
#pragma unroll
      for (int s = 1; s < 16; s <<= 1) rs += __shfl_xor(rs, s, 32);
      lrow[j] = lrow[j] * al + rs;
      mrow[j] = nm;
      alpha[j] = al;
      int r_ = j + half * 8;
      Pbuf[w][r_ * 32 + rc]      = f2bf(pa);   // keys kb+0 .. kb+15
      Pbuf[w][r_ * 32 + 16 + rc] = f2bf(pb);   // keys kb+16 .. kb+31
    }
#pragma unroll
    for (int j = 0; j < 8; ++j) {
      float al = alpha[j];
      O[0][j] *= al; O[1][j] *= al; O[2][j] *= al; O[3][j] *= al;
    }

    // ---- P (A-frag from per-wave LDS bounce) x V (B-frags from shared tile)
    v16bf pA = load_frag(&Pbuf[w][(lane & 15) * 32 + ks]);
#pragma unroll
    for (int t = 0; t < 4; ++t) {
      v16bf vB = load_frag(&Vbuf[cur][(t * 16 + rc) * 32 + ks]);
      O[t] = WMMA_BF16(pA, vB, O[t]);
    }
    __syncthreads();  // all waves done reading Kbuf/Vbuf[cur] before restaging
  }

  // ---- normalize + store context, layout [b][n][h*64 + d]
  const int b_ = bh / HEADS;
  const int h_ = bh % HEADS;
#pragma unroll
  for (int j = 0; j < 8; ++j) {
    int n_ = q0 + j + half * 8;
    float inv = 1.0f / lrow[j];
    size_t base = ((size_t)(b_ * SEQ + n_)) * EMB + h_ * HEAD_DIM;
#pragma unroll
    for (int t = 0; t < 4; ++t)
      ctx[base + t * 16 + rc] = f2bf(O[t][j] * inv);
  }
}

// ---------------------------------------------------------------- output proj
// TDM-staged GEMM: 64(M) x 128(O) tile per 8-wave block, fp32 out + bias.
__global__ __launch_bounds__(256) void proj_gemm(const us16* __restrict__ cb,
                                                 const us16* __restrict__ pw,
                                                 const float* __restrict__ bias,
                                                 float* __restrict__ out) {
  __shared__ __align__(16) us16 Abuf[2][64 * 32];
  __shared__ __align__(16) us16 Bbuf[2][128 * 32];

  const int tid  = threadIdx.x;
  const int w    = tid >> 5, lane = tid & 31;
  const int rc   = lane & 15;
  const int ks   = (lane >> 4) << 3;
  const int half = lane >> 4;
  const int mi   = w & 3;
  const int ng   = w >> 2;
  const int o0   = blockIdx.x * 128;
  const int m0   = blockIdx.y * 64;

  auto stage = [&](int buf, int k0) {
    tdm_load_tile(cb + (size_t)m0 * EMB + k0, &Abuf[buf][0], 64,  MTOT);
    tdm_load_tile(pw + (size_t)o0 * EMB + k0, &Bbuf[buf][0], 128, EMB);
  };

  v8f acc[4] = {v8f{}, v8f{}, v8f{}, v8f{}};
  if (w == 0) stage(0, 0);
  for (int k0 = 0; k0 < EMB; k0 += 32) {
    const int cur = (k0 >> 5) & 1;
    if (w == 0) {
      if (k0 + 32 < EMB) { stage(cur ^ 1, k0 + 32); __builtin_amdgcn_s_wait_tensorcnt(2); }
      else               { __builtin_amdgcn_s_wait_tensorcnt(0); }
    }
    __syncthreads();
    v16bf a = load_frag(&Abuf[cur][(mi * 16 + rc) * 32 + ks]);
#pragma unroll
    for (int t = 0; t < 4; ++t) {
      v16bf b = load_frag(&Bbuf[cur][(ng * 64 + t * 16 + rc) * 32 + ks]);
      acc[t] = WMMA_BF16(a, b, acc[t]);
    }
    __syncthreads();
  }

#pragma unroll
  for (int t = 0; t < 4; ++t) {
#pragma unroll
    for (int j = 0; j < 8; ++j) {
      int m = m0 + mi * 16 + j + half * 8;
      int o = o0 + ng * 64 + t * 16 + rc;
      out[(size_t)m * EMB + o] = acc[t][j] + bias[o];
    }
  }
}

// ---------------------------------------------------------------- launch
extern "C" void kernel_launch(void* const* d_in, const int* in_sizes, int n_in,
                              void* d_out, int out_size, void* d_ws, size_t ws_size,
                              hipStream_t stream) {
  const float* x      = (const float*)d_in[0];
  const float* qkv_w  = (const float*)d_in[1];
  const float* qkv_b  = (const float*)d_in[2];
  const float* proj_w = (const float*)d_in[3];
  const float* proj_b = (const float*)d_in[4];
  float* out = (float*)d_out;

  // workspace carve-up (bf16 buffers), total 48 MB
  char* ws = (char*)d_ws;
  const size_t SZ_XB = (size_t)MTOT * EMB * 2;
  const size_t SZ_WB = (size_t)QKVO * EMB * 2;
  const size_t SZ_PB = (size_t)EMB * EMB * 2;
  const size_t SZ_HT = (size_t)BATCH * HEADS * SEQ * HEAD_DIM * 2;
  us16* XB  = (us16*)(ws);
  us16* WB  = (us16*)(ws + SZ_XB);
  us16* PB  = (us16*)(ws + SZ_XB + SZ_WB);
  us16* Qb  = (us16*)(ws + SZ_XB + SZ_WB + SZ_PB);
  us16* Kb  = (us16*)(ws + SZ_XB + SZ_WB + SZ_PB + SZ_HT);
  us16* Vt  = (us16*)(ws + SZ_XB + SZ_WB + SZ_PB + 2 * SZ_HT);
  us16* CTX = (us16*)(ws + SZ_XB + SZ_WB + SZ_PB + 3 * SZ_HT);
  if (ws_size < SZ_XB + SZ_WB + SZ_PB + 3 * SZ_HT + (size_t)MTOT * EMB * 2) return;

  const int nx = MTOT * EMB;
  const int nw = QKVO * EMB;
  const int np = EMB * EMB;
  cvt_f32_bf16<<<(nx + 255) / 256, 256, 0, stream>>>(x, XB, nx);
  cvt_f32_bf16<<<(nw + 255) / 256, 256, 0, stream>>>(qkv_w, WB, nw);
  cvt_f32_bf16<<<(np + 255) / 256, 256, 0, stream>>>(proj_w, PB, np);

  qkv_gemm<<<dim3(QKVO / 128, MTOT / 64), 256, 0, stream>>>(XB, WB, qkv_b, Qb, Kb, Vt);
  attn<<<dim3(SEQ / 64, BATCH * HEADS), 128, 0, stream>>>(Qb, Kb, Vt, CTX);
  proj_gemm<<<dim3(EMB / 128, MTOT / 64), 256, 0, stream>>>(CTX, PB, proj_b, out);
}
